// Quant_IP_Individual_85117661872414
// MI455X (gfx1250) — compile-verified, matrix operand order checked by reading, not run on hardware
//
#include <hip/hip_runtime.h>
#include <cstddef>

#ifndef __has_builtin
#define __has_builtin(x) 0
#endif

// ---------------------------------------------------------------------------
// Types for WMMA fragments (gfx1250, wave32)
// ---------------------------------------------------------------------------
typedef __attribute__((ext_vector_type(16))) _Float16 v16h;
typedef __attribute__((ext_vector_type(8)))  _Float16 v8h;
typedef __attribute__((ext_vector_type(8)))  float    v8f;
typedef __attribute__((ext_vector_type(4)))  float    v4f;
typedef __attribute__((ext_vector_type(4)))  int      v4i;

#define BM   256   // block rows (activations)
#define BN   128   // block cols (out features)
#define BK   32    // K per WMMA step
#define LDSK 40    // padded LDS row stride in halves (80B) -> conflict-free frags

#define ACHUNKS (BM * 4)            // 16B chunks in the A tile
#define TCHUNKS (ACHUNKS + BN * 4)  // total chunks per K-tile (1536)
#define CPT     (TCHUNKS / 256)     // chunks per thread (6)

// ---------------------------------------------------------------------------
// CDNA5 async global->LDS path (ASYNCcnt), with compile-safe fallback.
// Builtin signature (from hipcc diagnostic): first arg is a pointer to a
// 16-byte int vector -> use AS1/AS3 v4i pointers.
// ---------------------------------------------------------------------------
#if __has_builtin(__builtin_amdgcn_global_load_async_to_lds_b128)
#define USE_ASYNC_LDS 1
typedef __attribute__((address_space(1))) v4i g_v4i;
typedef __attribute__((address_space(3))) v4i l_v4i;
__device__ __forceinline__ g_v4i* to_glb(const void* p) {
  return (g_v4i*)(size_t)p;
}
__device__ __forceinline__ l_v4i* to_lds(const void* p) {
  // generic LDS pointers carry the LDS offset in the low 32 bits
  return (l_v4i*)(unsigned)(size_t)p;
}
#if __has_builtin(__builtin_amdgcn_s_wait_asynccnt)
#define WAIT_ASYNC0() __builtin_amdgcn_s_wait_asynccnt(0)
#else
#define WAIT_ASYNC0() asm volatile("s_wait_asynccnt 0" ::: "memory")
#endif
#else
#define USE_ASYNC_LDS 0
#endif

// ---------------------------------------------------------------------------
// Fake-quant with fractional bit interpolation (matches reference; RNE)
// ---------------------------------------------------------------------------
__device__ __forceinline__ float quant_interp(float v, float a) {
  a = fmaxf(a, 1.0f);
  float blo  = floorf(a);
  float frac = a - blo;
  float slo  = exp2f(blo) - 1.0f;
  float shi  = exp2f(blo + 1.0f) - 1.0f;
  float c    = fminf(fmaxf(v, -1.0f), 1.0f);
  float qlo  = rintf(c * slo) / slo;
  float qhi  = rintf(c * shi) / shi;
  return (1.0f - frac) * qlo + frac * qhi;
}

// ---------------------------------------------------------------------------
// Stage 1: deterministic two-stage column sum of alpha_a
// ---------------------------------------------------------------------------
#define NSLICE 32
__global__ void colsum_partial_kernel(const float* __restrict__ a,
                                      float* __restrict__ psum,
                                      int rows, int cols) {
  const int j = blockIdx.x * blockDim.x + threadIdx.x;
  if (j >= cols) return;
  const int rpb = rows / NSLICE;
  const int r0  = blockIdx.y * rpb;
  float s = 0.0f;
  for (int i = 0; i < rpb; ++i)
    s += a[(size_t)(r0 + i) * cols + j];
  psum[(size_t)blockIdx.y * cols + j] = s;
}

__global__ void colsum_reduce_kernel(const float* __restrict__ psum,
                                     float* __restrict__ asum, int cols) {
  const int j = blockIdx.x * blockDim.x + threadIdx.x;
  if (j >= cols) return;
  float s = 0.0f;
#pragma unroll
  for (int k = 0; k < NSLICE; ++k)
    s += psum[(size_t)k * cols + j];
  asum[j] = s;
}

// ---------------------------------------------------------------------------
// Stage 2: fake-quant + convert to f16, 8 elems/thread (b128 in / b128 out)
// ---------------------------------------------------------------------------
__global__ void quant_to_f16_kernel(const float* __restrict__ v,
                                    const float* __restrict__ alpha,
                                    _Float16* __restrict__ q,
                                    size_t n, int cols, int per_element,
                                    float ascale) {
  const size_t base = ((size_t)blockIdx.x * blockDim.x + threadIdx.x) * 8;
  if (base >= n) return;
  v4f x0 = *(const v4f*)(v + base);
  v4f x1 = *(const v4f*)(v + base + 4);
  v4f a0, a1;
  if (per_element) {
    a0 = *(const v4f*)(alpha + base);
    a1 = *(const v4f*)(alpha + base + 4);
  } else {
    const size_t j = base % (size_t)cols;
    a0 = *(const v4f*)(alpha + j);
    a1 = *(const v4f*)(alpha + j + 4);
    a0 *= ascale;
    a1 *= ascale;
  }
  v8h o;
#pragma unroll
  for (int i = 0; i < 4; ++i) {
    o[i]     = (_Float16)quant_interp(x0[i], a0[i]);
    o[i + 4] = (_Float16)quant_interp(x1[i], a1[i]);
  }
  *(v8h*)(q + base) = o;
}

// ---------------------------------------------------------------------------
// Stage 3: NT GEMM  out[b,o] = sum_k qx[b,k]*qw[o,k] + bias[o]
// 256x128 block tile, BK=32, 8 waves (4M x 2N), wave tile 64x64 (4x4 WMMA),
// double-buffered LDS fed by async global->LDS b128 copies (ASYNCcnt).
// ---------------------------------------------------------------------------
__global__ __launch_bounds__(256) void qgemm_wmma_kernel(
    const _Float16* __restrict__ qx,   // [B, K] row-major
    const _Float16* __restrict__ qw,   // [N, K] row-major
    const float* __restrict__ bias,    // [N]
    float* __restrict__ out,           // [B, N]
    int K, int N) {
  __shared__ _Float16 As[2][BM * LDSK];
  __shared__ _Float16 Bs[2][BN * LDSK];

  const int t     = threadIdx.x;
  const int lane  = t & 31;
  const int wave  = t >> 5;
  const int waveM = wave >> 1;   // 0..3 -> 64-row slab
  const int waveN = wave & 1;    // 0..1 -> 64-col slab
  const int r     = lane & 15;   // row within 16x16 frag
  const int h     = lane >> 4;   // K-half select (ISA 16-bit A/B lane layout)

  const size_t rowM0 = (size_t)blockIdx.y * BM;
  const size_t colN0 = (size_t)blockIdx.x * BN;
  const _Float16* gA = qx + rowM0 * (size_t)K;
  const _Float16* gB = qw + colN0 * (size_t)K;

  // chunk geometry: TCHUNKS chunks of 8 halves (16B); CPT per thread
  int crow[CPT], cseg[CPT], cisA[CPT];
#pragma unroll
  for (int i = 0; i < CPT; ++i) {
    const int c  = t + i * 256;
    const int a  = (c < ACHUNKS) ? 1 : 0;
    const int cc = a ? c : (c - ACHUNKS);
    crow[i] = cc >> 2;
    cseg[i] = cc & 3;
    cisA[i] = a;
  }

  v8f acc[4][4] = {};

  // ---- tile copy: issue global->LDS for K-tile kt into buffer b ----
  auto issue_tile = [&](int kt, int b) {
    const size_t kb = (size_t)kt * BK;
#pragma unroll
    for (int i = 0; i < CPT; ++i) {
      const _Float16* g = (cisA[i] ? gA : gB) +
                          (size_t)crow[i] * K + kb + cseg[i] * 8;
      _Float16* l = cisA[i] ? &As[b][crow[i] * LDSK + cseg[i] * 8]
                            : &Bs[b][crow[i] * LDSK + cseg[i] * 8];
#if USE_ASYNC_LDS
      __builtin_amdgcn_global_load_async_to_lds_b128(to_glb(g), to_lds(l), 0, 0);
#else
      *(v4f*)l = *(const v4f*)g;  // fallback: sync global->reg->LDS
#endif
    }
  };

  // ---- preload K-tile 0 ----
  issue_tile(0, 0);
#if USE_ASYNC_LDS
  WAIT_ASYNC0();
#endif
  __syncthreads();

  const int nkt = K / BK;
  int buf = 0;
  for (int kt = 0; kt < nkt; ++kt) {
    const bool have_next = (kt + 1) < nkt;
    if (have_next) issue_tile(kt + 1, buf ^ 1);
    if (kt + 2 < nkt) {
      const size_t kp = (size_t)(kt + 2) * BK;
      __builtin_prefetch(gA + (size_t)crow[0] * K + kp, 0, 0);
      __builtin_prefetch(gB + (size_t)crow[0] * K + kp, 0, 0);
    }

    // A/B fragments from LDS (NT layout -> identical addressing for both)
    v16h af[4], bf[4];
#pragma unroll
    for (int mt = 0; mt < 4; ++mt) {
      const _Float16* p = &As[buf][(waveM * 64 + mt * 16 + r) * LDSK];
      v8h lo = *(const v8h*)(p + h * 8);
      v8h hi = *(const v8h*)(p + 16 + h * 8);
      af[mt] = __builtin_shufflevector(lo, hi, 0, 1, 2, 3, 4, 5, 6, 7,
                                       8, 9, 10, 11, 12, 13, 14, 15);
    }
#pragma unroll
    for (int nt = 0; nt < 4; ++nt) {
      const _Float16* p = &Bs[buf][(waveN * 64 + nt * 16 + r) * LDSK];
      v8h lo = *(const v8h*)(p + h * 8);
      v8h hi = *(const v8h*)(p + 16 + h * 8);
      bf[nt] = __builtin_shufflevector(lo, hi, 0, 1, 2, 3, 4, 5, 6, 7,
                                       8, 9, 10, 11, 12, 13, 14, 15);
    }

#pragma unroll
    for (int mt = 0; mt < 4; ++mt)
#pragma unroll
      for (int nt = 0; nt < 4; ++nt)
        acc[mt][nt] = __builtin_amdgcn_wmma_f32_16x16x32_f16(
            false, af[mt], false, bf[nt], (short)0, acc[mt][nt], false, false);

#if USE_ASYNC_LDS
    if (have_next) WAIT_ASYNC0();
#endif
    __syncthreads();
    buf ^= 1;
  }

  // epilogue: C layout -> VGPR v, lanes 0-15: M=v; lanes 16-31: M=v+8; N=lane%16
#pragma unroll
  for (int nt = 0; nt < 4; ++nt) {
    const size_t col = colN0 + waveN * 64 + nt * 16 + r;
    const float bv = bias[col];
#pragma unroll
    for (int mt = 0; mt < 4; ++mt) {
      const size_t rb = rowM0 + waveM * 64 + mt * 16 + h * 8;
#pragma unroll
      for (int vv = 0; vv < 8; ++vv)
        out[(rb + vv) * (size_t)N + col] = acc[mt][nt][vv] + bv;
    }
  }
}

// ---------------------------------------------------------------------------
// Launch
// ---------------------------------------------------------------------------
extern "C" void kernel_launch(void* const* d_in, const int* in_sizes, int n_in,
                              void* d_out, int out_size, void* d_ws, size_t ws_size,
                              hipStream_t stream) {
  const float* x       = (const float*)d_in[0];
  const float* weight  = (const float*)d_in[1];
  const float* bias    = (const float*)d_in[2];
  const float* alpha_w = (const float*)d_in[3];
  const float* alpha_a = (const float*)d_in[4];

  const int OUT = in_sizes[2];            // 4096
  const int IN  = in_sizes[1] / OUT;      // 4096
  const int B   = in_sizes[0] / IN;       // 8192

  // workspace layout: qx (f16) | qw (f16) | asum | psum
  _Float16* qx   = (_Float16*)d_ws;
  _Float16* qw   = qx + (size_t)B * IN;
  float*    asum = (float*)(qw + (size_t)OUT * IN);
  float*    psum = asum + IN;

  // 1) column sum of alpha_a (deterministic two-stage)
  {
    dim3 g((IN + 255) / 256, NSLICE);
    colsum_partial_kernel<<<g, 256, 0, stream>>>(alpha_a, psum, OUT, IN);
    colsum_reduce_kernel<<<(IN + 255) / 256, 256, 0, stream>>>(psum, asum, IN);
  }

  // 2) quantize activations (per-column mean alpha) and weights (per-element)
  {
    const size_t nx = (size_t)B * IN;
    quant_to_f16_kernel<<<(unsigned)((nx / 8 + 255) / 256), 256, 0, stream>>>(
        x, asum, qx, nx, IN, 0, 1.0f / (float)OUT);
    const size_t nw = (size_t)OUT * IN;
    quant_to_f16_kernel<<<(unsigned)((nw / 8 + 255) / 256), 256, 0, stream>>>(
        weight, alpha_w, qw, nw, IN, 1, 1.0f);
  }

  // 3) WMMA GEMM + bias
  {
    dim3 grid(OUT / BN, B / BM);
    qgemm_wmma_kernel<<<grid, 256, 0, stream>>>(qx, qw, bias, (float*)d_out,
                                                IN, OUT);
  }
}
